// PairwiseDualDabCNNMultiViT2_62027917689465
// MI455X (gfx1250) — matrix operand order repaired and checked by reading, not compiled, not run on hardware
//
#include <hip/hip_runtime.h>
#include <hip/hip_bf16.h>

// ---------------------------------------------------------------------------
// Types for CDNA5 WMMA (wave32). v_wmma_f32_16x16x32_bf16: D = A(16x32) * B(32x16) + C
// ---------------------------------------------------------------------------
typedef __bf16 bf16;
typedef __attribute__((ext_vector_type(16))) __bf16 bf16x16;
typedef __attribute__((ext_vector_type(8)))  __bf16 bf16x8;
typedef __attribute__((ext_vector_type(8)))  float  f32x8;

// A-matrix 16x32 bf16 per-lane layout (ISA 7.12.2): lane m (0..15) holds row m,
// K = [0..7] and [16..23]; lane m+16 holds row m, K = [8..15] and [24..31].
__device__ __forceinline__ bf16x16 load_a_frag(const bf16* rowptr, int kbase) {
    bf16x8 lo = *reinterpret_cast<const bf16x8*>(rowptr + kbase);
    bf16x8 hi = *reinterpret_cast<const bf16x8*>(rowptr + kbase + 16);
    bf16x16 r;
#pragma unroll
    for (int i = 0; i < 8; ++i) { r[i] = lo[i]; r[8 + i] = hi[i]; }
    return r;
}

// B-matrix 32x16 bf16: lane n (0..15) holds column n, K = [0..7] and [16..23];
// lane n+16 holds column n, K = [8..15] and [24..31]. Column n of B == row n of W (C = A * W^T).
__device__ __forceinline__ bf16x16 load_b_frag(const bf16* rowptr, int koff) {
    bf16x8 lo = *reinterpret_cast<const bf16x8*>(rowptr + koff);
    bf16x8 hi = *reinterpret_cast<const bf16x8*>(rowptr + koff + 16);
    bf16x16 r;
#pragma unroll
    for (int i = 0; i < 8; ++i) { r[i] = lo[i]; r[8 + i] = hi[i]; }
    return r;
}

// ---------------------------------------------------------------------------
// Generic batched GEMM:  C[m,n] = scale * ( sum_k A[m,k]*W[n,k] + bias[n] )
// A: M x K (row stride lda), W: N x K (row stride ldw)  -> C = A * W^T
// Block: 256 threads = 8 waves; block tile 128(M) x 128(N); wave tile 32 x 64.
// Requires M%128==0, N%128==0, K%32==0 (true for every call below).
// ---------------------------------------------------------------------------
__global__ __launch_bounds__(256)
void gemm_bf16_wmma(const bf16* __restrict__ A, long long lda, long long sAb,
                    const bf16* __restrict__ W, long long ldw, long long sWb,
                    const float* __restrict__ bias,
                    float* __restrict__ Cf, long long ldcf, long long sCfb,
                    bf16* __restrict__ Cb, long long ldcb, long long sCbb,
                    float scale, int M, int N, int K)
{
    const long long bz = blockIdx.z;
    A += bz * sAb;
    W += bz * sWb;
    if (Cf) Cf += bz * sCfb;
    if (Cb) Cb += bz * sCbb;

    const int lane  = threadIdx.x & 31;
    const int wid   = threadIdx.x >> 5;
    const int waveM = wid & 3;           // 4 waves along M
    const int waveN = wid >> 2;          // 2 waves along N
    const int mBase = blockIdx.y * 128 + waveM * 32;
    const int nBase = blockIdx.x * 128 + waveN * 64;

    const int rsel = lane & 15;
    const int aK   = (lane >> 4) * 8;    // A chunk base within 32-wide K step
    const int bK   = (lane >> 4) * 8;    // B chunk base within 32-wide K step

    f32x8 acc[2][4];
#pragma unroll
    for (int i = 0; i < 2; ++i)
#pragma unroll
        for (int j = 0; j < 4; ++j) acc[i][j] = f32x8{};

    const bf16* aRow0 = A + (long long)(mBase + rsel) * lda;
    const bf16* aRow1 = A + (long long)(mBase + 16 + rsel) * lda;
    const bf16* wRow[4];
#pragma unroll
    for (int j = 0; j < 4; ++j)
        wRow[j] = W + (long long)(nBase + j * 16 + rsel) * ldw;

    for (int k0 = 0; k0 < K; k0 += 32) {
        bf16x16 a0 = load_a_frag(aRow0 + k0, aK);
        bf16x16 a1 = load_a_frag(aRow1 + k0, aK);
        bf16x16 b[4];
#pragma unroll
        for (int j = 0; j < 4; ++j) b[j] = load_b_frag(wRow[j] + k0, bK);
#pragma unroll
        for (int j = 0; j < 4; ++j)
            acc[0][j] = __builtin_amdgcn_wmma_f32_16x16x32_bf16(
                false, a0, false, b[j], (short)0, acc[0][j], false, false);
#pragma unroll
        for (int j = 0; j < 4; ++j)
            acc[1][j] = __builtin_amdgcn_wmma_f32_16x16x32_bf16(
                false, a1, false, b[j], (short)0, acc[1][j], false, false);
    }

    // Epilogue. C/D layout: VGPR v -> M = mBase+i*16 + v + (lane>=16 ? 8:0), N = nBase+j*16+(lane&15)
    float bv[4];
#pragma unroll
    for (int j = 0; j < 4; ++j)
        bv[j] = bias ? bias[nBase + j * 16 + rsel] : 0.0f;
    const int mOff = (lane >> 4) * 8;
#pragma unroll
    for (int i = 0; i < 2; ++i) {
#pragma unroll
        for (int j = 0; j < 4; ++j) {
            const int n = nBase + j * 16 + rsel;
#pragma unroll
            for (int v = 0; v < 8; ++v) {
                const int m = mBase + i * 16 + mOff + v;
                const float val = (acc[i][j][v] + bv[j]) * scale;
                if (Cf) Cf[(long long)m * ldcf + n] = val;
                if (Cb) Cb[(long long)m * ldcb + n] = (bf16)val;
            }
        }
    }
}

// ---------------------------------------------------------------------------
// Small prologue / epilogue kernels
// ---------------------------------------------------------------------------
// Channel pool: per pixel, max & mean over C=256 channels of ifreq (NCHW).
__global__ __launch_bounds__(256)
void k_pool(const float* __restrict__ ifreq, float* __restrict__ cmax,
            float* __restrict__ cmean)
{
    int idx = blockIdx.x * 256 + threadIdx.x;           // B*H*W = 65536
    int b = idx >> 12, hw = idx & 4095;
    const float* p = ifreq + ((long long)b * 256) * 4096 + hw;
    float mx = -3.4e38f, sm = 0.f;
    for (int c = 0; c < 256; ++c) {
        float v = p[(long long)c * 4096];
        mx = fmaxf(mx, v);
        sm += v;
    }
    cmax[idx]  = mx;
    cmean[idx] = sm * (1.0f / 256.0f);
}

// 3x3 SAME conv over [max,mean] -> BN -> sigmoid gate (one scalar per pixel).
__global__ __launch_bounds__(256)
void k_gate(const float* __restrict__ cmax, const float* __restrict__ cmean,
            const float* __restrict__ cw, const float* __restrict__ gamma,
            const float* __restrict__ beta, const float* __restrict__ mean,
            const float* __restrict__ var, float* __restrict__ gate)
{
    int idx = blockIdx.x * 256 + threadIdx.x;           // 65536
    int b = idx >> 12, hw = idx & 4095;
    int h = hw >> 6, w = hw & 63;
    float s = 0.f;
    for (int ch = 0; ch < 2; ++ch) {
        const float* src = ch ? cmean : cmax;
        for (int ky = 0; ky < 3; ++ky) {
            int hh = h + ky - 1;
            if (hh < 0 || hh > 63) continue;
            for (int kx = 0; kx < 3; ++kx) {
                int ww = w + kx - 1;
                if (ww < 0 || ww > 63) continue;
                s += cw[(ch * 3 + ky) * 3 + kx] * src[b * 4096 + hh * 64 + ww];
            }
        }
    }
    float bn = (s - mean[0]) * rsqrtf(var[0] + 1e-5f) * gamma[0] + beta[0];
    gate[idx] = 1.0f / (1.0f + expf(-bn));
}

// Per-channel global average: y[b,c] = mean over HW. One block per (b,c).
__global__ __launch_bounds__(256)
void k_chan_mean(const float* __restrict__ ifreq, float* __restrict__ y)
{
    __shared__ float sh[256];
    int bc = blockIdx.x, t = threadIdx.x;
    const float* p = ifreq + (long long)bc * 4096;
    float s = 0.f;
    for (int i = 0; i < 16; ++i) s += p[t + i * 256];
    sh[t] = s; __syncthreads();
    for (int o = 128; o > 0; o >>= 1) {
        if (t < o) sh[t] += sh[t + o];
        __syncthreads();
    }
    if (t == 0) y[bc] = sh[0] * (1.0f / 4096.0f);
}

// CALayer MLP: aw = sigmoid(relu(y*W1^T+b1)*W2^T+b2). One block per batch.
__global__ __launch_bounds__(256)
void k_ca_mlp(const float* __restrict__ y, const float* __restrict__ w1,
              const float* __restrict__ b1, const float* __restrict__ w2,
              const float* __restrict__ b2, float* __restrict__ aw)
{
    __shared__ float hc[16];
    int b = blockIdx.x, t = threadIdx.x;
    if (t < 16) {
        float s = b1[t];
        for (int c = 0; c < 256; ++c) s += y[b * 256 + c] * w1[t * 256 + c];
        hc[t] = fmaxf(s, 0.f);
    }
    __syncthreads();
    float s = b2[t];
    for (int j = 0; j < 16; ++j) s += hc[j] * w2[t * 16 + j];
    aw[b * 256 + t] = 1.0f / (1.0f + expf(-s));
}

// Generic f32 -> bf16 convert.
__global__ __launch_bounds__(256)
void k_f32_to_bf16(const float* __restrict__ in, bf16* __restrict__ out, long long n)
{
    long long i = (long long)blockIdx.x * 256 + threadIdx.x;
    if (i < n) out[i] = (bf16)in[i];
}

// Build pixel-major [g*ifreq | aw*ifreq] (65536 x 512, bf16).
// topk_rate==1.0 => gather is identity permutation, so ca == aw*ifreq directly.
__global__ __launch_bounds__(256)
void k_build_concat1(const float* __restrict__ ifreq, const float* __restrict__ gate,
                     const float* __restrict__ aw, bf16* __restrict__ out)
{
    long long idx = (long long)blockIdx.x * 256 + threadIdx.x;   // 16*256*4096
    long long b = idx >> 20;
    int c = (int)((idx >> 12) & 255), hw = (int)(idx & 4095);
    float v = ifreq[idx];
    long long p = (b << 12) + hw;
    out[p * 512 + c]       = (bf16)(v * gate[p]);
    out[p * 512 + 256 + c] = (bf16)(v * aw[(b << 8) + c]);
}

// rgb (NCHW f32) -> left half of pixel-major concat2 (65536 x 512 bf16).
__global__ __launch_bounds__(256)
void k_build_rgbhalf(const float* __restrict__ rgb, bf16* __restrict__ out)
{
    long long idx = (long long)blockIdx.x * 256 + threadIdx.x;   // 16*256*4096
    long long b = idx >> 20;
    int c = (int)((idx >> 12) & 255), hw = (int)(idx & 4095);
    long long p = (b << 12) + hw;
    out[p * 512 + c] = (bf16)rgb[idx];
}

// Patchify dab (pixel-major bf16, 65536x256) into token-major rows inside Fused
// (row stride 8192): pure row gather, fully coalesced in c.
__global__ __launch_bounds__(256)
void k_patchify_dab(const bf16* __restrict__ dab, bf16* __restrict__ fused)
{
    long long idx = (long long)blockIdx.x * 256 + threadIdx.x;   // 16*256*16*256
    int c  = (int)(idx & 255);
    int pp = (int)((idx >> 8) & 15);
    int n  = (int)((idx >> 12) & 255);
    long long b = idx >> 20;
    int h = (n >> 4) * 4 + (pp >> 2);
    int w = (n & 15) * 4 + (pp & 3);
    long long src = ((b << 12) + h * 64 + w) * 256 + c;
    long long dst = ((b << 8) + n) * 8192 + pp * 256 + c;
    fused[dst] = dab[src];
}

// Patchify freq (NCHW f32) -> Xf (token-major 4096x4096 bf16) and
// XfT (per batch 4096x256 bf16, XfT[b,d,n] = Xf[b,n,d]) for S @ Xf.
__global__ __launch_bounds__(256)
void k_patchify_freq(const float* __restrict__ freq, bf16* __restrict__ Xf,
                     bf16* __restrict__ XfT)
{
    long long idx = (long long)blockIdx.x * 256 + threadIdx.x;   // 16*256*16*256
    int c  = (int)(idx & 255);
    int pp = (int)((idx >> 8) & 15);
    int n  = (int)((idx >> 12) & 255);
    long long b = idx >> 20;
    int h = (n >> 4) * 4 + (pp >> 2);
    int w = (n & 15) * 4 + (pp & 3);
    float v = freq[((b << 8) + c) * 4096 + h * 64 + w];
    int d = pp * 256 + c;
    Xf[((b << 8) + n) * 4096 + d]        = (bf16)v;
    XfT[((b << 12) + d) * 256 + n]       = (bf16)v;
}

// Row softmax over 256 logits -> bf16. One block per row.
__global__ __launch_bounds__(256)
void k_softmax256(const float* __restrict__ S, bf16* __restrict__ Sb)
{
    __shared__ float sh[256];
    __shared__ float red[2];
    int row = blockIdx.x, t = threadIdx.x;
    float v = S[(long long)row * 256 + t];
    sh[t] = v; __syncthreads();
    for (int o = 128; o > 0; o >>= 1) {
        if (t < o) sh[t] = fmaxf(sh[t], sh[t + o]);
        __syncthreads();
    }
    if (t == 0) red[0] = sh[0];
    __syncthreads();
    float e = expf(v - red[0]);
    sh[t] = e; __syncthreads();
    for (int o = 128; o > 0; o >>= 1) {
        if (t < o) sh[t] += sh[t + o];
        __syncthreads();
    }
    if (t == 0) red[1] = 1.0f / sh[0];
    __syncthreads();
    Sb[(long long)row * 256 + t] = (bf16)(e * red[1]);
}

// ---------------------------------------------------------------------------
// Host orchestration
// ---------------------------------------------------------------------------
extern "C" void kernel_launch(void* const* d_in, const int* in_sizes, int n_in,
                              void* d_out, int out_size, void* d_ws, size_t ws_size,
                              hipStream_t stream)
{
    (void)in_sizes; (void)n_in; (void)out_size; (void)ws_size;
    // B=16, C=256, H=W=64, P=4, N=256 tokens, D=4096, DIM=1024
    const float* ifreq    = (const float*)d_in[0];
    const float* rgb      = (const float*)d_in[1];
    const float* freq     = (const float*)d_in[2];
    const float* sa_w     = (const float*)d_in[3];
    const float* sa_gamma = (const float*)d_in[4];
    const float* sa_beta  = (const float*)d_in[5];
    const float* sa_mean  = (const float*)d_in[6];
    const float* sa_var   = (const float*)d_in[7];
    const float* ca_w1    = (const float*)d_in[8];
    const float* ca_b1    = (const float*)d_in[9];
    const float* ca_w2    = (const float*)d_in[10];
    const float* ca_b2    = (const float*)d_in[11];
    const float* c11_w    = (const float*)d_in[12];
    const float* c11_b    = (const float*)d_in[13];
    const float* c13_w    = (const float*)d_in[14];
    const float* c13_b    = (const float*)d_in[15];
    const float* wq       = (const float*)d_in[16];
    const float* wk       = (const float*)d_in[17];
    /* d_in[18] == wv : dead in the reference (v discarded) */
    const float* wo       = (const float*)d_in[19];
    const float* bo       = (const float*)d_in[20];
    const float* emb_w    = (const float*)d_in[21];
    const float* emb_b    = (const float*)d_in[22];
    float* out            = (float*)d_out;

    char* ws = (char*)d_ws;
    size_t off = 0;
    auto take = [&](size_t bytes) -> char* {
        char* p = ws + off;
        off += (bytes + 255) & ~(size_t)255;
        return p;
    };
    // small scalars/maps
    float* cmax  = (float*)take(65536 * 4);
    float* cmean = (float*)take(65536 * 4);
    float* gate  = (float*)take(65536 * 4);
    float* yvec  = (float*)take(4096 * 4);
    float* awv   = (float*)take(4096 * 4);
    // bf16 weights
    bf16* c11b = (bf16*)take((size_t)256 * 512 * 2);
    bf16* c13b = (bf16*)take((size_t)256 * 512 * 2);
    bf16* wqb  = (bf16*)take((size_t)4096 * 4096 * 2);
    bf16* wkb  = (bf16*)take((size_t)4096 * 4096 * 2);
    bf16* wob  = (bf16*)take((size_t)4096 * 4096 * 2);
    bf16* embb = (bf16*)take((size_t)1024 * 8192 * 2);
    // big activation buffers (aliased across phases)
    bf16* concat1 = (bf16*)take((size_t)65536 * 512 * 2);   // later: Qb | Kb
    bf16* concat2 = (bf16*)take((size_t)65536 * 512 * 2);   // later: Xf | XfT
    bf16* dab_pm  = (bf16*)take((size_t)65536 * 256 * 2);   // later: Afreq
    bf16* fused   = (bf16*)take((size_t)4096 * 8192 * 2);   // [Xr | 0.5*O]
    float* Sf     = (float*)take((size_t)16 * 256 * 256 * 4);
    bf16*  Sb     = (bf16*)take((size_t)16 * 256 * 256 * 2);
    // aliases
    bf16* Qb    = concat1;                               // 4096 x 4096
    bf16* Kb    = concat1 + (size_t)4096 * 4096;         // 4096 x 4096
    bf16* Xf    = concat2;                               // 4096 x 4096 (token-major)
    bf16* XfT   = concat2 + (size_t)4096 * 4096;         // 16 x (4096 x 256)
    bf16* Afreq = dab_pm;                                // 4096 x 4096

    auto gemm = [&](const bf16* A, long long lda, long long sAb,
                    const bf16* W, long long ldw, long long sWb,
                    const float* bias, float* Cf, long long ldcf, long long sCf,
                    bf16* Cb, long long ldcb, long long sCb,
                    float scale, int M, int N, int K, int batch) {
        dim3 grid((unsigned)(N / 128), (unsigned)(M / 128), (unsigned)batch);
        gemm_bf16_wmma<<<grid, 256, 0, stream>>>(A, lda, sAb, W, ldw, sWb, bias,
                                                 Cf, ldcf, sCf, Cb, ldcb, sCb,
                                                 scale, M, N, K);
    };

    // ---- DAB prologue -----------------------------------------------------
    k_pool<<<256, 256, 0, stream>>>(ifreq, cmax, cmean);
    k_gate<<<256, 256, 0, stream>>>(cmax, cmean, sa_w, sa_gamma, sa_beta,
                                    sa_mean, sa_var, gate);
    k_chan_mean<<<4096, 256, 0, stream>>>(ifreq, yvec);
    k_ca_mlp<<<16, 256, 0, stream>>>(yvec, ca_w1, ca_b1, ca_w2, ca_b2, awv);

    // ---- weight conversions to bf16 --------------------------------------
    auto conv = [&](const float* in, bf16* o, long long n) {
        k_f32_to_bf16<<<(unsigned)((n + 255) / 256), 256, 0, stream>>>(in, o, n);
    };
    conv(c11_w, c11b, 256 * 512);
    conv(c13_w, c13b, 256 * 512);
    conv(wq, wqb, (long long)4096 * 4096);
    conv(wk, wkb, (long long)4096 * 4096);
    conv(wo, wob, (long long)4096 * 4096);
    conv(emb_w, embb, (long long)1024 * 8192);

    // ---- DAB fuse as two GEMMs (pixel-major) ------------------------------
    k_build_concat1<<<65536, 256, 0, stream>>>(ifreq, gate, awv, concat1);
    k_build_rgbhalf<<<65536, 256, 0, stream>>>(rgb, concat2);
    // attn = [g*ifreq | aw*ifreq] @ c11_w^T + c11_b  -> right half of concat2
    gemm(concat1, 512, 0, c11b, 512, 0, c11_b,
         nullptr, 0, 0, concat2 + 256, 512, 0, 1.0f, 65536, 256, 512, 1);
    // dab = [rgb | attn] @ c13_w^T + c13_b  -> pixel-major bf16
    gemm(concat2, 512, 0, c13b, 512, 0, c13_b,
         nullptr, 0, 0, dab_pm, 256, 0, 1.0f, 65536, 256, 512, 1);

    // ---- patchify ---------------------------------------------------------
    k_patchify_dab<<<65536, 256, 0, stream>>>(dab_pm, fused);      // Xr -> fused left (ld 8192)
    k_patchify_freq<<<65536, 256, 0, stream>>>(freq, Xf, XfT);     // (concat2 now free; reuse)

    // ---- attention --------------------------------------------------------
    // Q = (Xr @ wq^T) / 64  (1/sqrt(4096) folded into bf16 store)
    gemm(fused, 8192, 0, wqb, 4096, 0, nullptr,
         nullptr, 0, 0, Qb, 4096, 0, 1.0f / 64.0f, 4096, 4096, 4096, 1);
    // K = Xf @ wk^T
    gemm(Xf, 4096, 0, wkb, 4096, 0, nullptr,
         nullptr, 0, 0, Kb, 4096, 0, 1.0f, 4096, 4096, 4096, 1);
    // S = Q @ K^T   (per-batch 256x256, K-dim 4096)
    gemm(Qb, 4096, (long long)256 * 4096, Kb, 4096, (long long)256 * 4096, nullptr,
         Sf, 256, (long long)256 * 256, nullptr, 0, 0, 1.0f, 256, 256, 4096, 16);
    k_softmax256<<<4096, 256, 0, stream>>>(Sf, Sb);
    // Afreq = softmax(S) @ Xf   (per-batch; W = Xf^T, ldw = 256)
    gemm(Sb, 256, (long long)256 * 256, XfT, 256, (long long)4096 * 256, nullptr,
         nullptr, 0, 0, Afreq, 4096, (long long)256 * 4096, 1.0f, 256, 4096, 256, 16);
    // 0.5 * (Afreq @ wo^T + bo) -> right half of fused (ld 8192)
    gemm(Afreq, 4096, 0, wob, 4096, 0, bo,
         nullptr, 0, 0, fused + 4096, 8192, 0, 0.5f, 4096, 4096, 4096, 1);

    // ---- final embedding: [Xr | 0.5*out] @ emb_w^T + emb_b -> f32 d_out ---
    gemm(fused, 8192, 0, embb, 8192, 0, emb_b,
         out, 1024, 0, nullptr, 0, 0, 1.0f, 4096, 1024, 8192, 1);
}
